// LocalSelfAttentionMultiHeadConcat_32959579030110
// MI455X (gfx1250) — matrix operand order repaired and checked by reading, not compiled, not run on hardware
//
#include <hip/hip_runtime.h>

// Problem constants (from reference)
#define Bn   2
#define Cc   512
#define Tt   32
#define HWp  196
#define Nh   8
#define CIc  64
#define KWk  9
#define TPp  40              // T + 2*P, P=4
#define TOK  (Bn * Tt * HWp) // 12544 tokens (multiple of 16)

typedef __attribute__((ext_vector_type(16))) __bf16 v16bf;
typedef __attribute__((ext_vector_type(8)))  __bf16 v8bf;
typedef __attribute__((ext_vector_type(8)))  float  v8f;
typedef __attribute__((ext_vector_type(4)))  unsigned int v4u;
typedef __attribute__((ext_vector_type(8)))  unsigned int v8u;

struct __attribute__((aligned(8))) bh8a8 { __bf16 v[8]; };

__device__ __forceinline__ v8bf ld8(const __bf16* p) { return *(const v8bf*)p; }
__device__ __forceinline__ v8bf ld8a8(const __bf16* p) { // only 8B-aligned
  bh8a8 t = *(const bh8a8*)p;
  v8bf r;
  for (int i = 0; i < 8; ++i) r[i] = t.v[i];
  return r;
}
__device__ __forceinline__ v16bf cat16(v8bf a, v8bf b) {
  return __builtin_shufflevector(a, b, 0,1,2,3,4,5,6,7,8,9,10,11,12,13,14,15);
}
__device__ __forceinline__ v8f wmma_bf16(v16bf a, v16bf b, v8f c) {
  return __builtin_amdgcn_wmma_f32_16x16x32_bf16(false, a, false, b, (short)0, c,
                                                 false, false);
}

// Tensor Data Mover: 2D tile (tile_d1 lines of tile_d0 units, line stride stride_d0)
// from global -> LDS. D# groups built per CDNA5 ISA and
// handed to the TDM in SGPR tuples. Tracked with TENSORcnt.
__device__ __forceinline__ void tdm_load_2d(unsigned lds_addr, unsigned long long gaddr,
                                            unsigned data_size_log2, unsigned tile_d0,
                                            unsigned tile_d1, unsigned long long stride_d0,
                                            unsigned tensor_d0, unsigned tensor_d1) {
  v4u g0;
  g0[0] = 1u;                                                   // count=1, user mode
  g0[1] = lds_addr;                                             // LDS byte address
  g0[2] = (unsigned)gaddr;                                      // global_addr[31:0]
  g0[3] = ((unsigned)(gaddr >> 32) & 0x01FFFFFFu) | (2u << 30); // addr[56:32], type=2
  v8u g1;
  g1[0] = (data_size_log2 & 3u) << 16;              // wg_mask=0, data_size
  g1[1] = (tensor_d0 & 0xFFFFu) << 16;              // tensor_dim0[15:0]
  g1[2] = (tensor_d0 >> 16) | ((tensor_d1 & 0xFFFFu) << 16);
  g1[3] = (tensor_d1 >> 16) | (tile_d0 << 16);      // tile_dim0
  g1[4] = tile_d1 & 0xFFFFu;                        // tile_dim1 (tile_dim2=0)
  g1[5] = (unsigned)stride_d0;                      // tensor_dim0_stride[31:0]
  g1[6] = (unsigned)(stride_d0 >> 32) & 0xFFFFu;    // stride[47:32]
  g1[7] = 0u;
  asm volatile("tensor_load_to_lds %0, %1" :: "s"(g0), "s"(g1) : "memory");
}

// ---------------- layout-conversion kernels ----------------

// xb[token][c] bf16; token = (b*T + t)*HW + p
__global__ void cvt_x_kernel(const float* __restrict__ x, __bf16* __restrict__ xb) {
  long i = (long)blockIdx.x * blockDim.x + threadIdx.x;
  if (i >= (long)TOK * Cc) return;
  int c = (int)(i % Cc);
  int token = (int)(i / Cc);
  int b = token / (Tt * HWp);
  int rem = token % (Tt * HWp);
  int t = rem / HWp, p = rem % HWp;
  xb[i] = (__bf16)x[(((long)b * Cc + c) * Tt + t) * HWp + p];
}

// wt[which][c][oc] bf16 (c-major so B-fragments are contiguous over oc)
__global__ void cvt_w_kernel(const float* __restrict__ Wq, const float* __restrict__ Wk,
                             const float* __restrict__ Wv, __bf16* __restrict__ wt) {
  long i = (long)blockIdx.x * blockDim.x + threadIdx.x;
  if (i >= 3L * Cc * Cc) return;
  int which = (int)(i / (Cc * Cc));
  int rem = (int)(i % (Cc * Cc));
  int c = rem / Cc, oc = rem % Cc;
  const float* W = (which == 0) ? Wq : ((which == 1) ? Wk : Wv);
  wt[i] = (__bf16)W[(long)oc * Cc + c];
}

// Fill padded K/V frames with bias (reference pads x with zeros => k = bk, v = bv)
__global__ void init_kv_kernel(const float* __restrict__ bk, const float* __restrict__ bv,
                               __bf16* __restrict__ kT, __bf16* __restrict__ v) {
  long i = (long)blockIdx.x * blockDim.x + threadIdx.x;
  long total = (long)Bn * Nh * TPp * HWp * CIc;
  if (i >= total) return;
  int ci = (int)(i & 63);
  long r = i >> 6;
  int p = (int)(r % HWp); r /= HWp;
  int tp = (int)(r % TPp); r /= TPp;
  int bn = (int)r;            // b*Nh + n
  int n = bn & 7;
  v[i] = (__bf16)bv[n * CIc + ci];
  long kTi = ((long)bn * CIc + ci) * (TPp * HWp) + (long)tp * HWp + p;
  kT[kTi] = (__bf16)bk[n * CIc + ci];
}

// ---------------- QKV projection GEMM (WMMA bf16) ----------------
__global__ void proj_kernel(const __bf16* __restrict__ xb, const __bf16* __restrict__ wt,
                            const float* __restrict__ bq, const float* __restrict__ bk,
                            const float* __restrict__ bv,
                            __bf16* __restrict__ q_ws, __bf16* __restrict__ kT_ws,
                            __bf16* __restrict__ v_ws) {
  const int lane = threadIdx.x & 31;
  const int wv = threadIdx.x >> 5;
  const int l15 = lane & 15;
  const int h = lane >> 4;
  const int rt = blockIdx.x;
  const int octile = blockIdx.y * 4 + wv;
  const int which = blockIdx.z;              // 0=q 1=k 2=v
  const __bf16* W = wt + (long)which * Cc * Cc;
  const float* bias = (which == 0) ? bq : ((which == 1) ? bk : bv);

  const int arow = rt * 16 + l15;
  v8f acc = {};
#pragma unroll 4
  for (int kb = 0; kb < Cc; kb += 32) {
    const __bf16* xr = xb + (long)arow * Cc + kb;
    v16bf a = cat16(ld8(xr + 8 * h), ld8(xr + 16 + 8 * h));
    const __bf16* wr = W + (long)kb * Cc + octile * 16 + l15;
    v16bf bm;
    for (int i = 0; i < 16; ++i) bm[i] = wr[(long)((i & 7) + 8 * h + 16 * (i >> 3)) * Cc];
    acc = wmma_bf16(a, bm, acc);
  }
  const int oc = octile * 16 + l15;
  const int n = oc >> 6, ci = oc & 63;
  const float bb = bias[oc];
#pragma unroll
  for (int r = 0; r < 8; ++r) {
    int token = rt * 16 + r + 8 * h;
    float val = acc[r] + bb;
    int b = token / (Tt * HWp);
    int rem = token % (Tt * HWp);
    int t = rem / HWp, p = rem % HWp;
    int bn = b * Nh + n;
    if (which == 0) {
      q_ws[(((long)bn * Tt + t) * HWp + p) * CIc + ci] = (__bf16)val;
    } else if (which == 1) {
      kT_ws[((long)bn * CIc + ci) * (TPp * HWp) + (long)(t + 4) * HWp + p] = (__bf16)val;
    } else {
      v_ws[(((long)bn * TPp + (t + 4)) * HWp + p) * CIc + ci] = (__bf16)val;
    }
  }
}

// ---------------- flash attention over the 9-frame window ----------------
// One block (8 wave32) per (b,n,t). Two uniform passes: wave owns query-row tile
// wv, then wv+8 (13 real 16-row tiles cover 196 query rows; extras are dummies).
// K^T and V slabs are DMA'd into double-buffered LDS by the TDM, overlapped with
// the WMMA work of the previous slab.
__global__ void attn_kernel(const __bf16* __restrict__ q_ws, const __bf16* __restrict__ kT_ws,
                            const __bf16* __restrict__ v_ws, const float* __restrict__ x,
                            float* __restrict__ out) {
  __shared__ __bf16 Vs[2][HWp * CIc];   // [key][ci]   25088 B per buffer
  __shared__ __bf16 Ks[2][CIc * HWp];   // [ci][key]   25088 B per buffer
  __shared__ __bf16 Ps[8][16 * 32];     // per-wave P scratch (D->A layout bridge)
  const int lane = threadIdx.x & 31;
  const int wv = threadIdx.x >> 5;
  const int l15 = lane & 15;
  const int h = lane >> 4;
  const int bid = blockIdx.x;
  const int b = bid >> 8;
  const int n = (bid >> 5) & 7;
  const int t = bid & 31;
  const int bn = b * Nh + n;

  const __bf16* qb = q_ws + ((long)bn * Tt + t) * HWp * CIc;
  const __bf16* kTb = kT_ws + (long)bn * CIc * (TPp * HWp);
  const __bf16* vb = v_ws + (long)bn * TPp * HWp * CIc;
  __bf16* Pw = &Ps[wv][0];

#pragma unroll 1
  for (int pass = 0; pass < 2; ++pass) {
    const int ttr = wv + 8 * pass;
    const bool act = ttr < 13;
    const int tt = act ? ttr : 12;
    // ---- Q fragments (A layout: lane row = l15, K runs of 8) ----
    const int qrow = tt * 16 + l15;
    v16bf qA0 = {}, qA1 = {};
    if (qrow < HWp) {
      const __bf16* qr = qb + (long)qrow * CIc;
      qA0 = cat16(ld8(qr + 8 * h), ld8(qr + 16 + 8 * h));
      qA1 = cat16(ld8(qr + 32 + 8 * h), ld8(qr + 48 + 8 * h));
    }
    v8f Y[4] = {v8f{}, v8f{}, v8f{}, v8f{}};
    float rowm[8], rowl[8];
#pragma unroll
    for (int r = 0; r < 8; ++r) { rowm[r] = -3.0e38f; rowl[r] = 0.0f; }

#pragma unroll 1
    for (int s = 0; s < KWk; ++s) {
      __syncthreads(); // everyone done with buffer (s-1)&1 before it is overwritten
      if (threadIdx.x == 0) {
        if (s == 0) { // prime the pipeline: slab 0 -> buffer 0
          int sr0 = t * HWp;
          tdm_load_2d((unsigned)(unsigned long long)(void*)&Vs[0][0],
                      (unsigned long long)(const void*)(vb + (long)sr0 * CIc),
                      3u, (HWp * CIc * 2) / 8, 1u, (HWp * CIc * 2) / 8,
                      0x3FFFFFFFu, 1u);
          tdm_load_2d((unsigned)(unsigned long long)(void*)&Ks[0][0],
                      (unsigned long long)(const void*)(kTb + sr0),
                      0u, HWp * 2, CIc, (unsigned long long)(TPp * HWp * 2),
                      0x3FFFFFFFu, CIc);
        }
        if (s + 1 < KWk) { // issue slab s+1 into the other buffer
          int sr1 = (t + s + 1) * HWp;
          tdm_load_2d((unsigned)(unsigned long long)(void*)&Vs[(s + 1) & 1][0],
                      (unsigned long long)(const void*)(vb + (long)sr1 * CIc),
                      3u, (HWp * CIc * 2) / 8, 1u, (HWp * CIc * 2) / 8,
                      0x3FFFFFFFu, 1u);
          tdm_load_2d((unsigned)(unsigned long long)(void*)&Ks[(s + 1) & 1][0],
                      (unsigned long long)(const void*)(kTb + sr1),
                      0u, HWp * 2, CIc, (unsigned long long)(TPp * HWp * 2),
                      0x3FFFFFFFu, CIc);
          __builtin_amdgcn_s_wait_tensorcnt(2); // slab s complete; s+1 in flight
        } else {
          __builtin_amdgcn_s_wait_tensorcnt(0); // drain
        }
      }
      __syncthreads(); // slab s visible to all waves
      const __bf16* Vbuf = &Vs[s & 1][0];
      const __bf16* Kbuf = &Ks[s & 1][0];

#pragma unroll 1
      for (int ch = 0; ch < 7; ++ch) {
        const int kb32 = ch * 32;
        // ---- scores: S(16x32) = Q(16x64) * K^T(64x32), 4 WMMAs (B frags from LDS) ----
        const int key0 = min(kb32 + l15, HWp - 1), key1 = min(kb32 + 16 + l15, HWp - 1);
        v16bf kB0a, kB1a, kB0b, kB1b;
        for (int i = 0; i < 16; ++i) {
          int ci = (i & 7) + 8 * h + 16 * (i >> 3);
          kB0a[i] = Kbuf[ci * HWp + key0];
          kB1a[i] = Kbuf[(32 + ci) * HWp + key0];
          kB0b[i] = Kbuf[ci * HWp + key1];
          kB1b[i] = Kbuf[(32 + ci) * HWp + key1];
        }
        v8f S0 = {}, S1 = {};
        S0 = wmma_bf16(qA0, kB0a, S0);
        S0 = wmma_bf16(qA1, kB1a, S0);
        S1 = wmma_bf16(qA0, kB0b, S1);
        S1 = wmma_bf16(qA1, kB1b, S1);
        // mask keys past the 196-key slab (only last chunk)
        if (kb32 + l15 >= HWp) {
#pragma unroll
          for (int r = 0; r < 8; ++r) S0[r] = -3.0e38f;
        }
        if (kb32 + 16 + l15 >= HWp) {
#pragma unroll
          for (int r = 0; r < 8; ++r) S1[r] = -3.0e38f;
        }
        // ---- online softmax (row = r + 8h; N across 16 lanes of each half) ----
#pragma unroll
        for (int r = 0; r < 8; ++r) {
          float mx = fmaxf(S0[r], S1[r]);
          mx = fmaxf(mx, __shfl_xor(mx, 1));
          mx = fmaxf(mx, __shfl_xor(mx, 2));
          mx = fmaxf(mx, __shfl_xor(mx, 4));
          mx = fmaxf(mx, __shfl_xor(mx, 8));
          float nm = fmaxf(rowm[r], mx);
          float corr = __expf(rowm[r] - nm);
          rowm[r] = nm;
          float p0 = __expf(S0[r] - nm);
          float p1 = __expf(S1[r] - nm);
          S0[r] = p0; S1[r] = p1;
          float rs = p0 + p1;
          rs += __shfl_xor(rs, 1);
          rs += __shfl_xor(rs, 2);
          rs += __shfl_xor(rs, 4);
          rs += __shfl_xor(rs, 8);
          rowl[r] = rowl[r] * corr + rs;
          Y[0][r] *= corr; Y[1][r] *= corr; Y[2][r] *= corr; Y[3][r] *= corr;
        }
        // ---- D-layout P -> A-layout via per-wave LDS ----
#pragma unroll
        for (int r = 0; r < 8; ++r) {
          Pw[(r + 8 * h) * 32 + l15] = (__bf16)S0[r];
          Pw[(r + 8 * h) * 32 + 16 + l15] = (__bf16)S1[r];
        }
        asm volatile("s_wait_dscnt 0" ::: "memory");
        const __bf16* pr = Pw + l15 * 32;
        v16bf pA = cat16(ld8(pr + 8 * h), ld8(pr + 16 + 8 * h));
        // ---- Y(16x64) += P(16x32) * V(32x64), 4 WMMAs (B frags from LDS) ----
#pragma unroll
        for (int ct = 0; ct < 4; ++ct) {
          v16bf vB;
          for (int i = 0; i < 16; ++i) {
            int kk = kb32 + (i & 7) + 8 * h + 16 * (i >> 3);
            if (kk > HWp - 1) kk = HWp - 1; // masked keys have P==0
            vB[i] = Vbuf[kk * CIc + ct * 16 + l15];
          }
          Y[ct] = wmma_bf16(pA, vB, Y[ct]);
        }
      }
    }
    // ---- epilogue: normalize, add residual, store ----
    if (act) {
#pragma unroll
      for (int ct = 0; ct < 4; ++ct) {
        const int c = n * CIc + ct * 16 + l15;
#pragma unroll
        for (int r = 0; r < 8; ++r) {
          int p = tt * 16 + r + 8 * h;
          if (p < HWp) {
            long idx = (((long)b * Cc + c) * Tt + t) * HWp + p;
            out[idx] = Y[ct][r] / rowl[r] + x[idx];
          }
        }
      }
    }
  }
}

// ---------------- host launcher ----------------
extern "C" void kernel_launch(void* const* d_in, const int* in_sizes, int n_in,
                              void* d_out, int out_size, void* d_ws, size_t ws_size,
                              hipStream_t stream) {
  const float* x  = (const float*)d_in[0];
  const float* Wq = (const float*)d_in[1];
  const float* bq = (const float*)d_in[2];
  const float* Wk = (const float*)d_in[3];
  const float* bk = (const float*)d_in[4];
  const float* Wv = (const float*)d_in[5];
  const float* bv = (const float*)d_in[6];
  float* out = (float*)d_out;
  char* ws = (char*)d_ws;

  const size_t sz_xb = (size_t)TOK * Cc * 2;
  const size_t sz_wt = (size_t)3 * Cc * Cc * 2;
  const size_t sz_q  = (size_t)Bn * Nh * Tt * HWp * CIc * 2;
  const size_t sz_kv = (size_t)Bn * Nh * TPp * HWp * CIc * 2;
  const size_t o_xb = 0;
  const size_t o_wt = o_xb + sz_xb;
  const size_t o_q  = o_wt + sz_wt;
  const size_t o_kT = o_q + sz_q;
  const size_t o_v  = o_kT + sz_kv + 4096; // slack
  (void)in_sizes; (void)n_in; (void)out_size; (void)ws_size;

  __bf16* xb = (__bf16*)(ws + o_xb);
  __bf16* wt = (__bf16*)(ws + o_wt);
  __bf16* qw = (__bf16*)(ws + o_q);
  __bf16* kT = (__bf16*)(ws + o_kT);
  __bf16* vw = (__bf16*)(ws + o_v);

  {
    long nel = (long)TOK * Cc;
    cvt_x_kernel<<<(unsigned)((nel + 255) / 256), 256, 0, stream>>>(x, xb);
  }
  {
    long nel = 3L * Cc * Cc;
    cvt_w_kernel<<<(unsigned)((nel + 255) / 256), 256, 0, stream>>>(Wq, Wk, Wv, wt);
  }
  {
    long nel = (long)Bn * Nh * TPp * HWp * CIc;
    init_kv_kernel<<<(unsigned)((nel + 255) / 256), 256, 0, stream>>>(bk, bv, kT, vw);
  }
  proj_kernel<<<dim3(TOK / 16, Cc / 64, 3), 128, 0, stream>>>(xb, wt, bq, bk, bv,
                                                              qw, kT, vw);
  attn_kernel<<<Bn * Nh * Tt, 256, 0, stream>>>(qw, kT, vw, x, out);
}
